// SpikingSelfAttention_28587302322709
// MI455X (gfx1250) — matrix operand order repaired and checked by reading, not compiled, not run on hardware
//
#include <hip/hip_runtime.h>
#include <hip/hip_bf16.h>

typedef _Float16 half_t;
typedef __attribute__((ext_vector_type(16))) _Float16 v16h;
typedef __attribute__((ext_vector_type(8)))  _Float16 v8h;
typedef __attribute__((ext_vector_type(4)))  _Float16 v4h;
typedef __attribute__((ext_vector_type(8)))  float    v8f;
typedef __attribute__((ext_vector_type(4)))  float    v4f;

#define TT 4
#define NN 32
#define CC 512
#define LL 196
#define LPAD 224
#define HH 8
#define DD 64
#define O3 1536
#define EPSV 1e-5f
#define DECAY 0.5f
#define SCALE_ATTN 0.125f
#define XS 40   // LDS row stride (halves): 80 B -> conflict-free b128 reads, 16B aligned

static __device__ inline v16h mk16(v8h lo, v8h hi) {
  return __builtin_shufflevector(lo, hi, 0,1,2,3,4,5,6,7,8,9,10,11,12,13,14,15);
}
// v16h operand from two 16-byte chunks (halves[0..7] at p0, halves[8..15] at p1)
static __device__ inline v16h frag16(const half_t* p0, const half_t* p1) {
  return mk16(*(const v8h*)p0, *(const v8h*)p1);
}
static __device__ inline v8f wmma_f16(v16h a, v16h b, v8f c) {
  return __builtin_amdgcn_wmma_f32_16x16x32_f16(false, a, false, b, (short)0, c, false, false);
}
static __device__ inline v8f v8f_zero() {
  v8f r;
#pragma unroll
  for (int i = 0; i < 8; ++i) r[i] = 0.0f;
  return r;
}

// ---------------------------------------------------------------------------
// One-shot passes: weights f32->f16 (same layout); x f32 -> xT f16 transposed
// [T*N][LPAD][CC] with L-padding zeroed. Clamped indices (no exec-mask chains).
// ---------------------------------------------------------------------------
__global__ __launch_bounds__(256) void cvt_w(const float* __restrict__ w,
                                             half_t* __restrict__ w16, int n4) {
  int i = (blockIdx.x * 256 + threadIdx.x);
  if (i < n4) {
    v4f f = *(const v4f*)(w + (size_t)i * 4);
    v4h o;
#pragma unroll
    for (int j = 0; j < 4; ++j) o[j] = (half_t)f[j];
    *(v4h*)(w16 + (size_t)i * 4) = o;
  }
}

__global__ __launch_bounds__(256) void transpose_x(const float* __restrict__ x,
                                                   half_t* __restrict__ xT) {
  __shared__ __align__(16) half_t Ls[32][72];    // [l][c], 4.5 KB
  const int tid   = threadIdx.x;
  const int lTile = blockIdx.x * 32;   // 7
  const int cTile = blockIdx.y * 64;   // 8
  const int tn    = blockIdx.z;        // T*N = 128

  // load: 64 c-rows x 32 l, coalesced along l; clamp OOB index, select 0
  {
    const int cl = tid >> 2;           // 0..63
    const int l8 = (tid & 3) * 8;      // 0,8,16,24
    const float* xr = x + ((size_t)tn * CC + cTile + cl) * (size_t)LL + lTile;
#pragma unroll
    for (int j = 0; j < 8; ++j) {
      int l = lTile + l8 + j;
      int idx = (l < LL) ? (l8 + j) : 0;       // clamped (row start always valid)
      float v = xr[idx];
      if (l >= LL) v = 0.0f;
      Ls[l8 + j][cl] = (half_t)v;
    }
  }
  __syncthreads();
  // store: rows l, 64 contiguous c per row -> v8h stores
  {
    const int lr = tid >> 3;           // 0..31
    const int c8 = (tid & 7) * 8;      // 0..56
    v8h o = *(const v8h*)&Ls[lr][c8];
    *(v8h*)(xT + ((size_t)tn * LPAD + lTile + lr) * (size_t)CC + cTile + c8) = o;
  }
}

// ---------------------------------------------------------------------------
// Fused GEMM (O x 512) @ (512 x L) + BN + LIF(1.0) over T.
// 8 waves = 4(o) x 2(l); block tile 64 x 32; K-outer / T-inner, 4 accumulators.
// Input is the transposed f16 activation [T*N][LPAD][CC]: LDS staging is a pure
// copy done with double-buffered global_load_async_to_lds_b128 (ASYNCcnt),
// overlapping the DMA for step k+1 with the WMMAs of step k.
// QKV=true : q channels -> qT [T][N][LPAD][512] (v8h), k/v -> kv [T][N][1024][LPAD]
// QKV=false: f32 spikes -> out [T][N][512][196]
// ---------------------------------------------------------------------------
template<bool QKV>
__global__ __launch_bounds__(256) void gemm_bn_lif(
    const half_t* __restrict__ xT,
    const half_t* __restrict__ W16,   // [OTOT][512] f16
    const float* __restrict__ gamma, const float* __restrict__ beta,
    const float* __restrict__ mean,  const float* __restrict__ var,
    half_t* __restrict__ qT, half_t* __restrict__ kv, float* __restrict__ out)
{
  __shared__ __align__(16) half_t Xs[2][TT][32][XS];   // double-buffered, 20 KB

  const int tid  = threadIdx.x;
  const int lane = tid & 31;
  const int wv   = tid >> 5;
  const int wo   = wv & 3;
  const int wl   = wv >> 2;
  const int lTile = blockIdx.x * 32;
  const int oTile = blockIdx.y * 64;
  const int n     = blockIdx.z;

  const int hl   = lane >> 4;
  const int colN = lane & 15;
  const int oRow0 = oTile + wo * 16 + hl * 8;     // C/D rows oRow0..+7
  const int lCol  = lTile + wl * 16 + colN;       // C/D column

  v8f inv, shf;
#pragma unroll
  for (int r = 0; r < 8; ++r) {
    int o = oRow0 + r;
    float iv = gamma[o] * rsqrtf(var[o] + EPSV);
    inv[r] = iv;
    shf[r] = beta[o] - mean[o] * iv;
  }

  const half_t* Wrow = W16 + (size_t)(oTile + wo * 16 + (lane & 15)) * CC;
  const int aK = hl * 8;

  const v8f zero = v8f_zero();
  v8f acc[TT];
#pragma unroll
  for (int t = 0; t < TT; ++t) acc[t] = zero;

  const unsigned ldsBase = (unsigned)(size_t)&Xs[0][0][0][0];

  // Async-copy one K-step (4 t-tiles of 32l x 32c f16): 512 x 16B chunks / 256 thr
  auto issue = [&](int k) {
    const int kBase = (k & 15) * 32;
    const int buf   = k & 1;
#pragma unroll
    for (int u = 0; u < 2; ++u) {
      const int id  = tid + 256 * u;
      const int tt  = id >> 7;          // 0..3
      const int rem = id & 127;
      const int lr  = rem >> 2;         // 0..31
      const int ch  = rem & 3;          // 16B chunk in row
      const unsigned lo = ldsBase +
          (unsigned)((((((buf * TT + tt) * 32) + lr) * XS + ch * 8) * 2));
      const half_t* gp = xT + (((size_t)tt * NN + n) * LPAD + (lTile + lr)) * (size_t)CC
                            + kBase + ch * 8;
      asm volatile("global_load_async_to_lds_b128 %0, %1, off"
                   :: "v"(lo), "v"((unsigned long long)(size_t)gp) : "memory");
    }
  };

  issue(0);
  for (int k = 0; k < CC / 32; ++k) {
    if (k + 1 < CC / 32) {
      issue(k + 1);                                    // prefetch next step
      asm volatile("s_wait_asynccnt 0x2" ::: "memory");  // step k's pair retired
    } else {
      asm volatile("s_wait_asynccnt 0x0" ::: "memory");
    }
    __syncthreads();

    const half_t* ap = Wrow + k * 32 + aK;
    v16h a = frag16(ap, ap + 16);
#pragma unroll
    for (int t = 0; t < TT; ++t) {
      const half_t* bp = &Xs[k & 1][t][wl * 16 + colN][hl * 8];
      acc[t] = wmma_f16(a, frag16(bp, bp + 16), acc[t]);
    }
    __syncthreads();   // buffer free for issue(k+2)
  }

  // BN + LIF scan over T
  v8f mem = zero;
  const bool isQ = QKV && (oTile < 512);
#pragma unroll
  for (int t = 0; t < TT; ++t) {
    v8h pack;
#pragma unroll
    for (int r = 0; r < 8; ++r) {
      float y = acc[t][r] * inv[r] + shf[r];
      float m = mem[r] + (y - mem[r]) * DECAY;
      float s = (m >= 1.0f) ? 1.0f : 0.0f;
      mem[r] = m * (1.0f - s);
      if constexpr (QKV) {
        pack[r] = (half_t)((lCol < LL) ? s : 0.0f);   // zero L padding
      } else {
        if (lCol < LL)
          out[(((size_t)t * NN + n) * CC + (oRow0 + r)) * (size_t)LL + lCol] = s;
      }
    }
    if constexpr (QKV) {
      if (isQ) {
        *(v8h*)(qT + (((size_t)t * NN + n) * LPAD + lCol) * (size_t)512 + oRow0) = pack;
      } else {
#pragma unroll
        for (int r = 0; r < 8; ++r)
          kv[(((size_t)t * NN + n) * 1024 + (oRow0 - 512 + r)) * (size_t)LPAD + lCol] = pack[r];
      }
    }
  }
}

// ---------------------------------------------------------------------------
// Stage 2: per (n,h): vk = v.k^T (K = L, contiguous in kv layout), park vk in
// LDS (f16 exact), attn = vk.q * 0.125, LIF(0.5), per-tile membrane across t.
// q B-fragments straight from transposed global qT. Spikes stored transposed.
// ---------------------------------------------------------------------------
__global__ __launch_bounds__(256) void attn_lif(
    const half_t* __restrict__ qT, const half_t* __restrict__ kv,
    half_t* __restrict__ sattnT)
{
  __shared__ __align__(16) half_t VK[64][64];

  const int tid  = threadIdx.x;
  const int lane = tid & 31;
  const int wv   = tid >> 5;
  const int hl   = lane >> 4;
  const int colN = lane & 15;
  const int h = blockIdx.x;
  const int n = blockIdx.y;

  const v8f zero = v8f_zero();
  float mem[7][8];
#pragma unroll
  for (int i = 0; i < 7; ++i)
#pragma unroll
    for (int r = 0; r < 8; ++r) mem[i][r] = 0.0f;

  const int vkRowT = wv >> 2;    // 0/1 (second tile at +2)
  const int vkColT = wv & 3;
  const int aM = lane & 15;
  const int aK = hl * 8;

  for (int t = 0; t < TT; ++t) {
    const size_t bkv = ((size_t)t * NN + n) * (size_t)1024;
    const half_t* kb = kv + (bkv + (size_t)(h * DD)) * LPAD;
    const half_t* vb = kv + (bkv + (size_t)(512 + h * DD)) * LPAD;

    // ---- phase 1: vk = v . k^T ----
    v8f vka0 = zero, vka1 = zero;
    const int bCol = vkColT * 16 + colN;
#pragma unroll
    for (int ks = 0; ks < 7; ++ks) {
      const int kBase = ks * 32;
      const half_t* kp  = kb + (size_t)bCol * LPAD + kBase + hl * 8;
      const half_t* vp0 = vb + (size_t)(vkRowT * 16 + aM) * LPAD + kBase + aK;
      const half_t* vp1 = vb + (size_t)((vkRowT + 2) * 16 + aM) * LPAD + kBase + aK;
      v16h b = frag16(kp, kp + 16);
      vka0 = wmma_f16(frag16(vp0, vp0 + 16), b, vka0);
      vka1 = wmma_f16(frag16(vp1, vp1 + 16), b, vka1);
    }
    __syncthreads();   // previous t's VK consumers are done
#pragma unroll
    for (int r = 0; r < 8; ++r) {
      VK[vkRowT * 16 + hl * 8 + r][vkColT * 16 + colN]       = (half_t)vka0[r];
      VK[(vkRowT + 2) * 16 + hl * 8 + r][vkColT * 16 + colN] = (half_t)vka1[r];
    }
    __syncthreads();

    // ---- phase 2: attn = vk . q ----
    const half_t* qn = qT + ((size_t)t * NN + n) * (size_t)LPAD * 512;
#pragma unroll
    for (int i = 0; i < 7; ++i) {
      const int tIdx = wv + 8 * i;      // 0..55
      const int rT = tIdx & 3;          // d row-tile
      const int cT = tIdx >> 2;         // l col-tile (14)
      const int lCol = cT * 16 + colN;
      v8f acc = zero;
#pragma unroll
      for (int ks = 0; ks < 2; ++ks) {
        const half_t* ap = &VK[rT * 16 + aM][ks * 32 + aK];
        const half_t* bp = qn + (size_t)lCol * 512 + h * DD + ks * 32 + hl * 8;
        acc = wmma_f16(frag16(ap, ap + 16), frag16(bp, bp + 16), acc);
      }
      v8h pack;
#pragma unroll
      for (int r = 0; r < 8; ++r) {
        float y = acc[r] * SCALE_ATTN;
        float m = mem[i][r] + (y - mem[i][r]) * DECAY;
        float s = (m >= 0.5f) ? 1.0f : 0.0f;
        mem[i][r] = m * (1.0f - s);
        pack[r] = (half_t)s;            // pad-l columns see q==0 -> never spike
      }
      *(v8h*)(sattnT + (((size_t)t * NN + n) * LPAD + lCol) * (size_t)CC
              + h * DD + rT * 16 + hl * 8) = pack;
    }
    __syncthreads();
  }
}

// ---------------------------------------------------------------------------
extern "C" void kernel_launch(void* const* d_in, const int* in_sizes, int n_in,
                              void* d_out, int out_size, void* d_ws, size_t ws_size,
                              hipStream_t stream) {
  (void)in_sizes; (void)n_in; (void)out_size; (void)ws_size;

  const float* x_seq  = (const float*)d_in[0];
  const float* qkv_w  = (const float*)d_in[1];
  const float* qkv_g  = (const float*)d_in[2];
  const float* qkv_b  = (const float*)d_in[3];
  const float* qkv_m  = (const float*)d_in[4];
  const float* qkv_v  = (const float*)d_in[5];
  const float* proj_w = (const float*)d_in[6];
  const float* proj_g = (const float*)d_in[7];
  const float* proj_b = (const float*)d_in[8];
  const float* proj_m = (const float*)d_in[9];
  const float* proj_v = (const float*)d_in[10];

  half_t* xT     = (half_t*)d_ws;                           // [T][N][224][512]  29.4 MB
  half_t* qT     = xT     + (size_t)TT * NN * LPAD * 512;   // [T][N][224][512]  29.4 MB
  half_t* kvbuf  = qT     + (size_t)TT * NN * LPAD * 512;   // [T][N][1024][224] 58.7 MB
  half_t* sattnT = kvbuf  + (size_t)TT * NN * 1024 * LPAD;  // [T][N][224][512]  29.4 MB
  half_t* wq16   = sattnT + (size_t)TT * NN * LPAD * 512;   // [1536][512]        1.5 MB
  half_t* wp16   = wq16   + (size_t)O3 * CC;                // [512][512]         0.5 MB

  dim3 blk(256, 1, 1);

  cvt_w<<<dim3((O3 * CC / 4 + 255) / 256), blk, 0, stream>>>(qkv_w, wq16, O3 * CC / 4);
  cvt_w<<<dim3((CC * CC / 4 + 255) / 256), blk, 0, stream>>>(proj_w, wp16, CC * CC / 4);
  transpose_x<<<dim3(7, 8, TT * NN), blk, 0, stream>>>(x_seq, xT);

  gemm_bn_lif<true><<<dim3(7, 24, 32), blk, 0, stream>>>(
      xT, wq16, qkv_g, qkv_b, qkv_m, qkv_v, qT, kvbuf, nullptr);

  attn_lif<<<dim3(HH, NN, 1), blk, 0, stream>>>(qT, kvbuf, sattnT);

  gemm_bn_lif<false><<<dim3(7, 8, 32), blk, 0, stream>>>(
      sattnT, wp16, proj_g, proj_b, proj_m, proj_v, nullptr, nullptr, (float*)d_out);
}
